// Attention_41437844472470
// MI455X (gfx1250) — matrix operand order repaired and checked by reading, not compiled, not run on hardware
//
#include <hip/hip_runtime.h>
#include <cstdint>
#include <cstddef>

#define N_SEQ 2048
#define CDIM  2048
#define HD    128
#define NH    16

typedef __attribute__((ext_vector_type(8)))  __bf16 v8bf;
typedef __attribute__((ext_vector_type(16))) __bf16 v16bf;
typedef __attribute__((ext_vector_type(8)))  float  v8f;

union BF16x16 { v16bf v; v8bf h[2]; };

__device__ __forceinline__ v8f wmma_bf16(v16bf a, v16bf b, v8f c) {
  // (neg_a, A, neg_b, B, c_mod, C, reuse_a, reuse_b)
  return __builtin_amdgcn_wmma_f32_16x16x32_bf16(false, a, false, b, (short)0, c, false, false);
}

__device__ __forceinline__ v8f v8f_zero() {
  v8f z;
  #pragma unroll
  for (int i = 0; i < 8; i++) z[i] = 0.0f;
  return z;
}

__device__ __forceinline__ __bf16 to_bf16(float f) {
  unsigned u = __builtin_bit_cast(unsigned, f);
  unsigned r = (u + 0x7FFFu + ((u >> 16) & 1u)) >> 16;
  return __builtin_bit_cast(__bf16, (unsigned short)r);
}

__device__ __forceinline__ float redmax16(float v) {
  #pragma unroll
  for (int m = 1; m < 16; m <<= 1) v = fmaxf(v, __shfl_xor(v, m, 16));
  return v;
}
__device__ __forceinline__ float redsum16(float v) {
  #pragma unroll
  for (int m = 1; m < 16; m <<= 1) v += __shfl_xor(v, m, 16);
  return v;
}

// CDNA5 async copy: global -> LDS, 16 bytes per lane, tracked by ASYNCcnt.
__device__ __forceinline__ void async_b128(const __bf16* gsrc, __bf16* ldst) {
  unsigned loff = (unsigned)(uintptr_t)ldst;          // low 32 bits = LDS offset
  asm volatile("global_load_async_to_lds_b128 %0, %1, off"
               :: "v"(loff), "v"((unsigned long long)(uintptr_t)gsrc)
               : "memory");
}
__device__ __forceinline__ void wait_async0() {
  asm volatile("s_wait_asynccnt 0x0" ::: "memory");
}

// ---------------------------------------------------------------------------
// fp32 -> bf16 cast (row-major copy)
// ---------------------------------------------------------------------------
__global__ void __launch_bounds__(256)
cast_f32_bf16(const float* __restrict__ in, __bf16* __restrict__ out, int n) {
  int i = blockIdx.x * blockDim.x + threadIdx.x;
  int stride = gridDim.x * blockDim.x;
  for (; i < n; i += stride) out[i] = to_bf16(in[i]);
}

// ---------------------------------------------------------------------------
// fp32 [rows][cols] -> bf16 transposed [cols][rows], 64x64 LDS tiles
// ---------------------------------------------------------------------------
__global__ void __launch_bounds__(256)
transpose_cast_f32_bf16(const float* __restrict__ in, __bf16* __restrict__ out,
                        int rows, int cols) {
  __shared__ float tile[64][65];
  const int t = threadIdx.x;
  const int r0 = blockIdx.y * 64;
  const int c0 = blockIdx.x * 64;
  #pragma unroll
  for (int i = 0; i < 16; i++) {
    int idx = t + 256 * i;
    int r = idx >> 6, c = idx & 63;
    tile[r][c] = in[(size_t)(r0 + r) * cols + c0 + c];
  }
  __syncthreads();
  #pragma unroll
  for (int i = 0; i < 16; i++) {
    int idx = t + 256 * i;
    int r = idx >> 6, c = idx & 63;
    out[(size_t)(c0 + r) * rows + r0 + c] = to_bf16(tile[c][r]);
  }
}

// ---------------------------------------------------------------------------
// Tiled bf16 GEMM, B pre-transposed (BT[n][k]).  64x128 tile per WG,
// 256 threads = 8 waves (16x64 per wave), K-step 64, double-buffered LDS
// fed by async global->LDS copies, one barrier per K-step.
// mode==1: QKV + RoPE epilogue (V written transposed [B,H,hd,N])
// mode==0: plain GEMM, f32 output
// ---------------------------------------------------------------------------
__global__ void __launch_bounds__(256)
gemm_bf16_kernel(const __bf16* __restrict__ A, const __bf16* __restrict__ BT,
                 int mode,
                 __bf16* __restrict__ q_ws, __bf16* __restrict__ k_ws,
                 __bf16* __restrict__ v_ws, float* __restrict__ outf) {
  __shared__ __align__(16) char smem[49152];
  __bf16* ldsA = (__bf16*)smem;             // [2][64*64]   16 KB
  __bf16* ldsB = (__bf16*)(smem + 16384);   // [2][128*64]  32 KB
  float*  ldsC = (float*)smem;              // overlay (valid after main loop)

  const int t = threadIdx.x;
  const int lane = t & 31;
  const int wv = t >> 5;
  const int lg = lane >> 4;
  const int ln = lane & 15;
  const int m0 = blockIdx.x * 64;
  const int c0 = blockIdx.y * 128;
  const int rowbase = 16 * (wv & 3);
  const int colbase = 64 * (wv >> 2);

  v8f acc[4];
  #pragma unroll
  for (int i = 0; i < 4; i++) acc[i] = v8f_zero();

  const int ar = t >> 2, aq = t & 3;    // A: 64 rows x 4 chunks of 16 halves
  const int br = t >> 1, b2 = t & 1;    // B: 128 rows x 2 chunks of 32 halves

  auto stage = [&](int buf, int k0) {
    __bf16* dA = ldsA + buf * 4096 + ar * 64 + aq * 16;
    const __bf16* sA = A + (size_t)(m0 + ar) * CDIM + k0 + aq * 16;
    async_b128(sA, dA);
    async_b128(sA + 8, dA + 8);
    __bf16* dB = ldsB + buf * 8192 + br * 64 + b2 * 32;
    const __bf16* sB = BT + (size_t)(c0 + br) * CDIM + k0 + b2 * 32;
    #pragma unroll
    for (int j = 0; j < 4; j++) async_b128(sB + 8 * j, dB + 8 * j);
  };

  stage(0, 0);
  for (int kt = 0; kt < CDIM / 64; kt++) {
    wait_async0();
    __syncthreads();
    if (kt + 1 < CDIM / 64) stage((kt + 1) & 1, (kt + 1) * 64);

    const __bf16* bA = ldsA + (kt & 1) * 4096;
    const __bf16* bB = ldsB + (kt & 1) * 8192;
    #pragma unroll
    for (int ks = 0; ks < 2; ks++) {
      BF16x16 af;  // lane = row rowbase+ln; K chunks {8*lg.., 16+8*lg..}
      const __bf16* pa = bA + (rowbase + ln) * 64 + ks * 32 + 8 * lg;
      af.h[0] = *(const v8bf*)pa;
      af.h[1] = *(const v8bf*)(pa + 16);
      #pragma unroll
      for (int cb = 0; cb < 4; cb++) {
        BF16x16 bf;  // lane = column; K chunks {8*lg.., 16+8*lg..}
        const __bf16* pb = bB + (colbase + 16 * cb + ln) * 64 + ks * 32 + 8 * lg;
        bf.h[0] = *(const v8bf*)pb;
        bf.h[1] = *(const v8bf*)(pb + 16);
        acc[cb] = wmma_bf16(af.v, bf.v, acc[cb]);
      }
    }
  }
  __syncthreads();   // A/B buffers dead; safe to overlay ldsC

  #pragma unroll
  for (int cb = 0; cb < 4; cb++) {
    #pragma unroll
    for (int r = 0; r < 8; r++)
      ldsC[(rowbase + r + 8 * lg) * 128 + colbase + 16 * cb + ln] = acc[cb][r];
  }
  __syncthreads();

  if (mode == 0) {
    for (int i = t; i < 64 * 128; i += 256) {
      int row = i >> 7, d = i & 127;
      outf[(size_t)(m0 + row) * CDIM + c0 + d] = ldsC[i];
    }
  } else {
    const int qkv = blockIdx.y >> 4;
    const int h = blockIdx.y & 15;
    for (int i = t; i < 64 * 128; i += 256) {
      int row = i >> 7, d = i & 127;
      int m = m0 + row;
      int b = m >> 11, n = m & (N_SEQ - 1);
      float val = ldsC[i];
      if (qkv == 2) {
        // V stored transposed: [B,H,hd,N] so flash B-frags are contiguous
        v_ws[((size_t)(b * NH + h) * HD + d) * N_SEQ + n] = to_bf16(val);
      } else {
        int pidx = (d < 64) ? d + 64 : d - 64;
        float pv = ldsC[row * 128 + pidx];
        float rot = (d < 64) ? -pv : pv;
        int j = d & 63;
        float freq = (float)n * __expf(-0.14391157f * (float)j);
        float s, c;
        __sincosf(freq, &s, &c);
        float o = val * c + rot * s;
        size_t oidx = ((size_t)(b * NH + h) * N_SEQ + n) * HD + d;
        ((qkv == 0) ? q_ws : k_ws)[oidx] = to_bf16(o);
      }
    }
  }
}

// ---------------------------------------------------------------------------
// Causal flash attention. Grid (N/64, B*H), 128 threads = 4 waves, each wave
// owns 16 query rows x hd=128. K frags stream from global (row-major K is
// already WMMA B-layout for S=Q*K^T); V^T staged via async double-buffered LDS.
// ---------------------------------------------------------------------------
__global__ void __launch_bounds__(128)
flash_attn_kernel(const __bf16* __restrict__ Q, const __bf16* __restrict__ K,
                  const __bf16* __restrict__ VT, __bf16* __restrict__ Oout) {
  __shared__ __bf16 ldsV[2][HD * 32];   // col-major [hd][key], 2 buffers
  __shared__ __bf16 ldsP[4][16 * 32];   // per-wave P staging

  const int t = threadIdx.x;
  const int lane = t & 31;
  const int wv = t >> 5;
  const int lg = lane >> 4;
  const int ln = lane & 15;
  const int bh = blockIdx.y;
  const int q0 = blockIdx.x * 64;
  const int qw = q0 + 16 * wv;

  const __bf16* Qh  = Q  + (size_t)bh * N_SEQ * HD;
  const __bf16* Kh  = K  + (size_t)bh * N_SEQ * HD;
  const __bf16* VTh = VT + (size_t)bh * HD * N_SEQ;

  // preload Q A-fragments (4 K-steps of 32 over hd=128)
  BF16x16 qf[4];
  #pragma unroll
  for (int s = 0; s < 4; s++) {
    const __bf16* p = Qh + (size_t)(qw + ln) * HD + 32 * s + 8 * lg;
    qf[s].h[0] = *(const v8bf*)p;
    qf[s].h[1] = *(const v8bf*)(p + 16);
  }

  v8f o[8];
  #pragma unroll
  for (int i = 0; i < 8; i++) o[i] = v8f_zero();
  float mrow[8], srow[8];
  #pragma unroll
  for (int r = 0; r < 8; r++) { mrow[r] = -1.0e30f; srow[r] = 0.0f; }

  const float SCALE = 0.0883883476f;   // 128^-0.5
  const int nkb = (q0 + 64) >> 5;

  auto stageV = [&](int buf, int key0) {
    __bf16* d = &ldsV[buf][t * 32];                     // t = hd column
    const __bf16* s = VTh + (size_t)t * N_SEQ + key0;   // contiguous keys
    #pragma unroll
    for (int j = 0; j < 4; j++) async_b128(s + 8 * j, d + 8 * j);
  };

  stageV(0, 0);
  for (int kb = 0; kb < nkb; kb++) {
    const int key0 = kb * 32;
    wait_async0();
    __syncthreads();
    if (kb + 1 < nkb) stageV((kb + 1) & 1, (kb + 1) * 32);
    const __bf16* Vb = &ldsV[kb & 1][0];

    // S = Q * K^T (two 16-key column fragments)
    v8f sc[2];
    #pragma unroll
    for (int cf = 0; cf < 2; cf++) {
      sc[cf] = v8f_zero();
      #pragma unroll
      for (int s = 0; s < 4; s++) {
        BF16x16 kf;
        const __bf16* p = Kh + (size_t)(key0 + 16 * cf + ln) * HD + 32 * s + 8 * lg;
        kf.h[0] = *(const v8bf*)p;
        kf.h[1] = *(const v8bf*)(p + 16);
        sc[cf] = wmma_bf16(qf[s].v, kf.v, sc[cf]);
      }
    }

    // scale + causal mask
    #pragma unroll
    for (int cf = 0; cf < 2; cf++) {
      #pragma unroll
      for (int r = 0; r < 8; r++) {
        float v = sc[cf][r] * SCALE;
        int row = qw + r + 8 * lg;
        int col = key0 + 16 * cf + ln;
        if (col > row) v = -1.0e30f;
        sc[cf][r] = v;
      }
    }

    // online softmax
    #pragma unroll
    for (int r = 0; r < 8; r++) {
      float mloc = redmax16(fmaxf(sc[0][r], sc[1][r]));
      float mn = fmaxf(mrow[r], mloc);
      float alpha = __expf(mrow[r] - mn);
      float p0 = __expf(sc[0][r] - mn);
      float p1 = __expf(sc[1][r] - mn);
      sc[0][r] = p0;
      sc[1][r] = p1;
      float rs = redsum16(p0 + p1);
      srow[r] = srow[r] * alpha + rs;
      mrow[r] = mn;
      #pragma unroll
      for (int cb = 0; cb < 8; cb++) o[cb][r] *= alpha;
    }

    // P: C-layout -> A-layout via per-wave LDS round trip
    __bf16* pp = &ldsP[wv][0];
    #pragma unroll
    for (int cf = 0; cf < 2; cf++) {
      #pragma unroll
      for (int r = 0; r < 8; r++)
        pp[(r + 8 * lg) * 32 + 16 * cf + ln] = to_bf16(sc[cf][r]);
    }
    asm volatile("s_wait_dscnt 0x0" ::: "memory");
    BF16x16 pf;
    {
      const __bf16* pa = pp + ln * 32 + 8 * lg;
      pf.h[0] = *(const v8bf*)pa;
      pf.h[1] = *(const v8bf*)(pa + 16);
    }

    // O += P * V
    #pragma unroll
    for (int cb = 0; cb < 8; cb++) {
      BF16x16 vf;
      const __bf16* vb = Vb + (16 * cb + ln) * 32 + 8 * lg;
      vf.h[0] = *(const v8bf*)vb;
      vf.h[1] = *(const v8bf*)(vb + 16);
      o[cb] = wmma_bf16(pf.v, vf.v, o[cb]);
    }
  }

  // normalize, store to attn workspace [B, N, C] (col = h*128 + d)
  float inv[8];
  #pragma unroll
  for (int r = 0; r < 8; r++) inv[r] = 1.0f / srow[r];
  const int b = bh >> 4, h = bh & 15;
  #pragma unroll
  for (int cb = 0; cb < 8; cb++) {
    #pragma unroll
    for (int r = 0; r < 8; r++) {
      int row = qw + r + 8 * lg;
      int col = h * HD + 16 * cb + ln;
      Oout[(size_t)(b * N_SEQ + row) * CDIM + col] = to_bf16(o[cb][r] * inv[r]);
    }
  }
}

// ---------------------------------------------------------------------------
// Launch
// ---------------------------------------------------------------------------
extern "C" void kernel_launch(void* const* d_in, const int* in_sizes, int n_in,
                              void* d_out, int out_size, void* d_ws, size_t ws_size,
                              hipStream_t stream) {
  (void)in_sizes; (void)n_in; (void)out_size; (void)ws_size;
  const float* x     = (const float*)d_in[0];
  // d_in[1] = position_ids (arange tiled) -> recomputed as n in-kernel
  const float* Wqkv  = (const float*)d_in[2];
  const float* Wproj = (const float*)d_in[3];

  char* w = (char*)d_ws;
  __bf16* xh      = (__bf16*)(w);                   // 16 MB [4096 x 2048]
  __bf16* wqkvT   = (__bf16*)(w + 16777216ull);     // 24 MB [6144 x 2048] (transposed)
  __bf16* wprojT  = (__bf16*)(w + 41943040ull);     //  8 MB [2048 x 2048] (transposed)
  __bf16* q_ws    = (__bf16*)(w + 50331648ull);     // 16 MB [B,H,N,hd]
  __bf16* k_ws    = (__bf16*)(w + 67108864ull);     // 16 MB [B,H,N,hd]
  __bf16* vT_ws   = (__bf16*)(w + 83886080ull);     // 16 MB [B,H,hd,N]
  __bf16* attn    = (__bf16*)(w + 100663296ull);    // 16 MB [B,N,C]

  cast_f32_bf16<<<2048, 256, 0, stream>>>(x, xh, 2 * N_SEQ * CDIM);
  transpose_cast_f32_bf16<<<dim3(96, 32), 256, 0, stream>>>(Wqkv, wqkvT, CDIM, 3 * CDIM);
  transpose_cast_f32_bf16<<<dim3(32, 32), 256, 0, stream>>>(Wproj, wprojT, CDIM, CDIM);

  gemm_bf16_kernel<<<dim3(64, 48), 256, 0, stream>>>(
      xh, wqkvT, 1, q_ws, k_ws, vT_ws, nullptr);

  flash_attn_kernel<<<dim3(32, 32), 128, 0, stream>>>(q_ws, k_ws, vT_ws, attn);

  gemm_bf16_kernel<<<dim3(64, 16), 256, 0, stream>>>(
      attn, wprojT, 0, nullptr, nullptr, nullptr, (float*)d_out);
}
